// PyUnidirectionalSingleLayerLstm_83193516523857
// MI455X (gfx1250) — matrix operand order repaired and checked by reading, compiled, not run on hardware
//
#include <hip/hip_runtime.h>
#include <hip/hip_bf16.h>
#include <stdint.h>

// ---------------------------------------------------------------------------
// LSTMP (packed sequence) for gfx1250: bf16 WMMA 16x16x32, fp32 accumulate.
// Shared A operand staged in LDS via async global->LDS copies,
// weights streamed from L2 as per-wave B fragments.
// ---------------------------------------------------------------------------

typedef __attribute__((ext_vector_type(16))) __bf16        v16bf;
typedef __attribute__((ext_vector_type(8)))  float         v8f;
typedef __attribute__((ext_vector_type(4)))  unsigned int  u32x4;
typedef __attribute__((vector_size(4 * sizeof(int)))) int  v4i_ld;

#define CELL_CLIP 3.0f
#define PROJ_CLIP 3.0f

// LDS tile: 64 rows x 256 K, stride padded to 264 (528B = 132 dwords ->
// 4-bank shift per row -> conflict-free ds_load_b128 A fragments).
#define LDS_LD   264
#define KCHUNK   256

#if defined(__gfx1250__) && \
    __has_builtin(__builtin_amdgcn_global_load_async_to_lds_b128) && \
    __has_builtin(__builtin_amdgcn_s_wait_asynccnt)
#define USE_ASYNC_LDS 1
#else
#define USE_ASYNC_LDS 0
#endif

// round-to-nearest-even fp32 -> bf16
static __device__ __forceinline__ unsigned short f2bf(float f) {
    unsigned int u = __float_as_uint(f);
    u += 0x7FFFu + ((u >> 16) & 1u);
    return (unsigned short)(u >> 16);
}

static __device__ __forceinline__ float sigmoidf(float x) {
    return 1.0f / (1.0f + __expf(-x));
}

struct FragBits { u32x4 lo, hi; };
static __device__ __forceinline__ v16bf make_frag(u32x4 lo, u32x4 hi) {
    FragBits p{lo, hi};
    return __builtin_bit_cast(v16bf, p);
}

// stage 16 bytes global -> LDS (async direct-to-LDS path when available)
static __device__ __forceinline__ void stage16(unsigned short* dst_lds,
                                               const unsigned short* src) {
#if USE_ASYNC_LDS
    __builtin_amdgcn_global_load_async_to_lds_b128(
        (__attribute__((address_space(1))) v4i_ld*)(void*)src,
        (__attribute__((address_space(3))) v4i_ld*)(void*)dst_lds, 0, 0);
#else
    *(u32x4*)dst_lds = *(const u32x4*)src;
#endif
}

static __device__ __forceinline__ void stage_wait() {
#if USE_ASYNC_LDS
    __builtin_amdgcn_s_wait_asynccnt(0);
#endif
    __syncthreads();
}

// A fragment (16x32, MxK) served from the LDS tile (row-major, ld = LDS_LD).
// Lane L: m = L%16, half = L/16; elems [0..7] = K[8h..8h+8),
// elems [8..15] = K[16+8h..16+8h+8).
static __device__ __forceinline__ v16bf lds_a_frag(
        const unsigned short* sA, int row0, int k0, int lane) {
    int m = lane & 15, h = lane >> 4;
    const unsigned short* p = sA + (row0 + m) * LDS_LD + k0;
    u32x4 lo = *(const u32x4*)(p + 8 * h);
    u32x4 hi = *(const u32x4*)(p + 16 + 8 * h);
    return make_frag(lo, hi);
}

// B fragment (32x16, KxN) where B[k][n] = W[n][k], W row-major with ld.
// Lane L: n = L%16, half = L/16; elems [0..7] = K[8h..8h+8),
// elems [8..15] = K[16+8h..16+8h+8) from row n.
static __device__ __forceinline__ v16bf load_b_frag(
        const unsigned short* __restrict__ base, int nrow0,
        int ld, int k0, int lane) {
    int n = lane & 15, h = lane >> 4;
    const unsigned short* p = base + (size_t)(nrow0 + n) * ld + k0;
    u32x4 lo = *(const u32x4*)(p + 8 * h);
    u32x4 hi = *(const u32x4*)(p + 16 + 8 * h);
    return make_frag(lo, hi);
}

// ---------------------------------------------------------------------------
// Small setup kernels
// ---------------------------------------------------------------------------
__global__ void prep_starts_kernel(const int* __restrict__ bs,
                                   int* __restrict__ starts, int T) {
    if (threadIdx.x == 0 && blockIdx.x == 0) {
        int acc = 0;
        for (int i = 0; i < T; ++i) { starts[i] = acc; acc += bs[i]; }
    }
}

__global__ void cvt_bf16_kernel(const float* __restrict__ src,
                                unsigned short* __restrict__ dst, int n) {
    int i = blockIdx.x * blockDim.x + threadIdx.x;
    int stride = gridDim.x * blockDim.x;
    for (; i < n; i += stride) dst[i] = f2bf(src[i]);
}

__global__ void init_state_kernel(const float* __restrict__ h0,
                                  const float* __restrict__ c0,
                                  float* __restrict__ hf,
                                  unsigned short* __restrict__ hb,
                                  float* __restrict__ c, int nh, int nc) {
    int i = blockIdx.x * blockDim.x + threadIdx.x;
    int stride = gridDim.x * blockDim.x;
    int n = nh > nc ? nh : nc;
    for (; i < n; i += stride) {
        if (i < nh) { float v = h0[i]; hf[i] = v; hb[i] = f2bf(v); }
        if (i < nc) { c[i] = c0[i]; }
    }
}

__global__ void finalize_kernel(const float* __restrict__ hf,
                                const float* __restrict__ c,
                                float* __restrict__ out, int nh, int nc) {
    int i = blockIdx.x * blockDim.x + threadIdx.x;
    int stride = gridDim.x * blockDim.x;
    for (; i < nh + nc; i += stride)
        out[i] = (i < nh) ? hf[i] : c[i - nh];
}

// ---------------------------------------------------------------------------
// Per-step gates + cell kernel.
// Grid: 32 blocks x 128 threads (4 waves). Wave owns 16 columns of D_c and
// computes all 4 gates for all 64 batch rows -> gate fusion stays in regs.
// Activations (x_t then h) staged through LDS in 64x256 chunks; 4 phases of
// 8 WMMA K-steps each (K = 1024 total).
// ---------------------------------------------------------------------------
__global__ __launch_bounds__(128, 1)
void lstm_gates_kernel(const unsigned short* __restrict__ xbf,   // [total][512]
                       const unsigned short* __restrict__ wi,    // [8192][512]
                       const unsigned short* __restrict__ wh,    // [8192][512]
                       const float*          __restrict__ bias,  // [8192]
                       const unsigned short* __restrict__ hbf,   // [64][512]
                       float*                __restrict__ c,     // [64][2048]
                       unsigned short*       __restrict__ sbf,   // [64][2048]
                       const int* __restrict__ starts,
                       const int* __restrict__ bs,
                       int t, int total) {
    __shared__ unsigned short sA[64 * LDS_LD];   // 33 KB

    const int lane = threadIdx.x & 31;
    const int wave = threadIdx.x >> 5;
    const int n0   = (blockIdx.x * 4 + wave) * 16;   // column slice in D_c
    const int start = starts[t];
    const int bsz   = bs[t];
    const int ncol  = n0 + (lane & 15);
    const int half  = lane >> 4;

    // acc[gate][m-tile], bias-initialized (bias value is per-column).
    v8f acc[4][4];
#pragma unroll
    for (int g = 0; g < 4; ++g) {
        float bv = bias[g * 2048 + ncol];
#pragma unroll
        for (int mt = 0; mt < 4; ++mt)
#pragma unroll
            for (int e = 0; e < 8; ++e) acc[g][mt][e] = bv;
    }

    // 4 phases: x_t[K 0:256], x_t[K 256:512], h[K 0:256], h[K 256:512]
    for (int ph = 0; ph < 4; ++ph) {
        const unsigned short* src = (ph < 2) ? xbf : hbf;
        const unsigned short* W   = (ph < 2) ? wi : wh;
        const int kb = (ph & 1) * KCHUNK;
        const bool gather = (ph < 2);

        // cooperative stage: 64 rows x 32 chunks of 8 bf16
        for (int i = threadIdx.x; i < 64 * 32; i += 128) {
            int row = i >> 5;
            int c8  = (i & 31) << 3;
            int srow = row;
            if (gather) {
                srow = start + row;
                if (srow > total - 1) srow = total - 1;
            }
            stage16(&sA[row * LDS_LD + c8],
                    src + (size_t)srow * 512 + kb + c8);
        }
        stage_wait();

        for (int ks = 0; ks < 8; ++ks) {
            int k0 = ks * 32;
            v16bf a[4];
#pragma unroll
            for (int mt = 0; mt < 4; ++mt)
                a[mt] = lds_a_frag(sA, mt * 16, k0, lane);
#pragma unroll
            for (int g = 0; g < 4; ++g) {
                v16bf b = load_b_frag(W, g * 2048 + n0, 512, kb + k0, lane);
#pragma unroll
                for (int mt = 0; mt < 4; ++mt)
                    acc[g][mt] = __builtin_amdgcn_wmma_f32_16x16x32_bf16(
                        false, a[mt], false, b, (short)0, acc[g][mt],
                        false, false);
            }
        }
        __syncthreads();   // protect sA before next stage
    }

    // ---- fused gate nonlinearity, cell clip, masked c update ----
    // D-tile layout: lane holds column ncol, rows m = r + 8*half within tile.
#pragma unroll
    for (int mt = 0; mt < 4; ++mt) {
#pragma unroll
        for (int r = 0; r < 8; ++r) {
            int b = mt * 16 + half * 8 + r;
            float gi = acc[0][mt][r];
            float gf = acc[1][mt][r];
            float gc = acc[2][mt][r];
            float go = acc[3][mt][r];
            float cold = c[b * 2048 + ncol];
            float cn = sigmoidf(gi) * tanhf(gc) + sigmoidf(gf) * cold;
            cn = fminf(fmaxf(cn, -CELL_CLIP), CELL_CLIP);
            float s = sigmoidf(go) * tanhf(cn);
            if (b < bsz) c[b * 2048 + ncol] = cn;  // masked cell update
            sbf[b * 2048 + ncol] = f2bf(s);        // h-update masked later
        }
    }
}

// ---------------------------------------------------------------------------
// Per-step projection kernel: h' = clip(s @ w_proj^T).
// Grid: 8 blocks x 128 threads (4 waves); wave owns 16 of 512 output cols.
// s staged through LDS in 64x256 chunks; 8 phases x 8 K-steps (K = 2048).
// ---------------------------------------------------------------------------
__global__ __launch_bounds__(128, 1)
void lstm_proj_kernel(const unsigned short* __restrict__ sbf,  // [64][2048]
                      const unsigned short* __restrict__ wp,   // [512][2048]
                      float*                __restrict__ hf,   // [64][512]
                      unsigned short*       __restrict__ hbf,  // [64][512]
                      float*                __restrict__ outp, // packed [total][512]
                      const int* __restrict__ starts,
                      const int* __restrict__ bs,
                      int t) {
    __shared__ unsigned short sA[64 * LDS_LD];   // 33 KB

    const int lane = threadIdx.x & 31;
    const int wave = threadIdx.x >> 5;
    const int j0   = (blockIdx.x * 4 + wave) * 16;
    const int start = starts[t];
    const int bsz   = bs[t];
    const int j     = j0 + (lane & 15);
    const int half  = lane >> 4;

    v8f acc[4];
#pragma unroll
    for (int mt = 0; mt < 4; ++mt)
#pragma unroll
        for (int e = 0; e < 8; ++e) acc[mt][e] = 0.0f;

    for (int ph = 0; ph < 8; ++ph) {
        const int kb = ph * KCHUNK;

        for (int i = threadIdx.x; i < 64 * 32; i += 128) {
            int row = i >> 5;
            int c8  = (i & 31) << 3;
            stage16(&sA[row * LDS_LD + c8],
                    sbf + (size_t)row * 2048 + kb + c8);
        }
        stage_wait();

        for (int ks = 0; ks < 8; ++ks) {
            int k0 = ks * 32;
            v16bf b = load_b_frag(wp, j0, 2048, kb + k0, lane);
#pragma unroll
            for (int mt = 0; mt < 4; ++mt) {
                v16bf a = lds_a_frag(sA, mt * 16, k0, lane);
                acc[mt] = __builtin_amdgcn_wmma_f32_16x16x32_bf16(
                    false, a, false, b, (short)0, acc[mt], false, false);
            }
        }
        __syncthreads();
    }

#pragma unroll
    for (int mt = 0; mt < 4; ++mt) {
#pragma unroll
        for (int r = 0; r < 8; ++r) {
            int b = mt * 16 + half * 8 + r;
            if (b < bsz) {
                float v = fminf(fmaxf(acc[mt][r], -PROJ_CLIP), PROJ_CLIP);
                hf[b * 512 + j]  = v;          // fp32 hidden state
                hbf[b * 512 + j] = f2bf(v);    // bf16 copy for next GEMM
                outp[(size_t)(start + b) * 512 + j] = v;  // packed output
            }
        }
    }
}

// ---------------------------------------------------------------------------
// Host launcher
// ---------------------------------------------------------------------------
extern "C" void kernel_launch(void* const* d_in, const int* in_sizes, int n_in,
                              void* d_out, int out_size, void* d_ws, size_t ws_size,
                              hipStream_t stream) {
    const float* inputs   = (const float*)d_in[0];
    const float* init_h   = (const float*)d_in[1];
    const float* init_c   = (const float*)d_in[2];
    const float* w_input  = (const float*)d_in[3];
    const float* w_hidden = (const float*)d_in[4];
    const float* bias     = (const float*)d_in[5];
    const float* w_proj   = (const float*)d_in[6];
    const int*   bs       = (const int*)d_in[7];

    const int T     = in_sizes[7];
    const int Din   = 512, Dh = 512, Dc = 2048, G = 4 * Dc, B = 64;
    const int total = in_sizes[0] / Din;

    // workspace carve-up (256B aligned)
    char* ws = (char*)d_ws;
    size_t off = 0;
    auto carve = [&](size_t bytes) {
        size_t o = off;
        off = (off + bytes + 255) & ~(size_t)255;
        return o;
    };
    int*            starts = (int*)           (ws + carve((size_t)T * 4));
    unsigned short* xbf    = (unsigned short*)(ws + carve((size_t)total * Din * 2));
    unsigned short* wi     = (unsigned short*)(ws + carve((size_t)G * Din * 2));
    unsigned short* wh     = (unsigned short*)(ws + carve((size_t)G * Dh * 2));
    unsigned short* wp     = (unsigned short*)(ws + carve((size_t)Dh * Dc * 2));
    float*          hf     = (float*)         (ws + carve((size_t)B * Dh * 4));
    unsigned short* hb     = (unsigned short*)(ws + carve((size_t)B * Dh * 2));
    float*          cst    = (float*)         (ws + carve((size_t)B * Dc * 4));
    unsigned short* sbf    = (unsigned short*)(ws + carve((size_t)B * Dc * 2));

    // setup: prefix sums, bf16 conversions, state init (re-done every call)
    prep_starts_kernel<<<1, 32, 0, stream>>>(bs, starts, T);
    cvt_bf16_kernel<<<256, 256, 0, stream>>>(inputs,   xbf, total * Din);
    cvt_bf16_kernel<<<256, 256, 0, stream>>>(w_input,  wi,  G * Din);
    cvt_bf16_kernel<<<256, 256, 0, stream>>>(w_hidden, wh,  G * Dh);
    cvt_bf16_kernel<<<256, 256, 0, stream>>>(w_proj,   wp,  Dh * Dc);
    init_state_kernel<<<128, 256, 0, stream>>>(init_h, init_c, hf, hb, cst,
                                               B * Dh, B * Dc);

    // recurrence
    float* out_packed = (float*)d_out;
    for (int t = 0; t < T; ++t) {
        lstm_gates_kernel<<<32, 128, 0, stream>>>(xbf, wi, wh, bias, hb, cst,
                                                  sbf, starts, bs, t, total);
        lstm_proj_kernel<<<8, 128, 0, stream>>>(sbf, wp, hf, hb, out_packed,
                                                starts, bs, t);
    }

    // final (h_f, c_f) appended after packed outputs
    float* out_tail = (float*)d_out + (size_t)total * Dh;
    finalize_kernel<<<128, 256, 0, stream>>>(hf, cst, out_tail, B * Dh, B * Dc);
}
